// DeepTropNet_23295902613829
// MI455X (gfx1250) — matrix operand order repaired and checked by reading, not compiled, not run on hardware
//
#include <hip/hip_runtime.h>
#include <hip/hip_bf16.h>

// ---------------------------------------------------------------------------
// DeepTropNet forward for MI455X (gfx1250, wave32).
// Classical contractions -> v_wmma_f32_16x16x32_f16 (f32 accumulate).
// Tropical (max,+) contractions -> VALU v_max_num_f32 loops over LDS data.
// K staging in attention uses CDNA5 async global->LDS DMA (ASYNCcnt).
// ---------------------------------------------------------------------------

#define NLAYER 2
#define DIM    128
#define NH     8
#define DK     16
#define LTOK   197
#define FF     256
#define PP     8
#define NCLS   1000
#define BATCH  4
#define NPATCH 196
#define M_TOK  (BATCH * LTOK)   // 788
#define LPAD   208              // 13 * 16
#define VPAD   224              // 7 * 32
#define MT_L   13
#define EPSV   1e-5f

typedef __attribute__((ext_vector_type(16))) _Float16 v16h;
typedef __attribute__((ext_vector_type(8)))  float    v8f;

__device__ __forceinline__ v8f wmma_f16(v16h a, v16h b, v8f c) {
  return __builtin_amdgcn_wmma_f32_16x16x32_f16(
      /*neg_a=*/false, a, /*neg_b=*/false, b,
      /*c_mod=*/(short)0, c, /*reuse_a=*/false, /*reuse_b=*/false);
}

// ---------------------------------------------------------------------------
// Generic WMMA GEMM: Y[m,n] = epi(sum_k A[m,k] * W[n,k] + bias[n]) (+resid)
// A: [M,K] f32 row-major, W: [N,K] f32 row-major. K % 32 == 0, N % 16 == 0.
// One wave per 16x16 output tile. epi: 0=none, 1=gelu(exact), 2=sigmoid.
// ---------------------------------------------------------------------------
__global__ __launch_bounds__(32)
void gemm_wmma_kernel(const float* __restrict__ A, const float* __restrict__ W,
                      const float* __restrict__ bias, const float* __restrict__ resid,
                      float* __restrict__ Y, int M, int N, int K, int epi) {
  const int lane  = threadIdx.x;
  const int n0    = blockIdx.x * 16;
  const int m0    = blockIdx.y * 16;
  const int hl    = lane >> 4;      // half-wave select
  const int ll    = lane & 15;

  v8f acc = {};
  const int arow  = m0 + ll;
  const bool rowok = arow < M;
  const float* Arow = A + (size_t)arow * K;
  const float* Wrow = W + (size_t)(n0 + ll) * K;

  // Pull this tile's operand rows toward the WGP (global_prefetch_b8).
  if (rowok) __builtin_prefetch(Arow, 0, 0);
  __builtin_prefetch(Wrow, 0, 0);

  const int kt_count = K >> 5;
  for (int kt = 0; kt < kt_count; ++kt) {
    v16h af = {};
    v16h bf;
    if (rowok) {
      const int ka = kt * 32 + 8 * hl;   // ISA A layout: 8 consecutive, then +16
#pragma unroll
      for (int j = 0; j < 8; ++j) af[j]     = (_Float16)Arow[ka + j];
#pragma unroll
      for (int j = 0; j < 8; ++j) af[8 + j] = (_Float16)Arow[ka + 16 + j];
    }
    const int kb = kt * 32 + 8 * hl;     // ISA B layout: 8 consecutive, then +16
#pragma unroll
    for (int j = 0; j < 16; ++j) bf[j] = (_Float16)Wrow[kb + ((j < 8) ? j : 8 + j)];
    acc = wmma_f16(af, bf, acc);
  }

#pragma unroll
  for (int r = 0; r < 8; ++r) {
    const int m = m0 + r + 8 * hl;
    const int n = n0 + ll;
    if (m < M) {
      float v = acc[r] + bias[n];
      if (epi == 1)      v = 0.5f * v * (1.0f + erff(v * 0.70710678118654752f));
      else if (epi == 2) v = 1.0f / (1.0f + __expf(-v));
      if (resid) v += resid[(size_t)m * N + n];
      Y[(size_t)m * N + n] = v;
    }
  }
}

// ---------------------------------------------------------------------------
// Patch embed: h[b, 1+pi, :] = BN( xp @ patch_w.T + patch_b + pos[1+pi] )
// xp gathered from x on the fly. One wave per 16x16 tile, M=784, N=128, K=768.
// ---------------------------------------------------------------------------
__global__ __launch_bounds__(32)
void patch_embed_kernel(const float* __restrict__ x, const float* __restrict__ pw,
                        const float* __restrict__ pb, const float* __restrict__ pos,
                        const float* __restrict__ bn_g, const float* __restrict__ bn_b,
                        const float* __restrict__ bn_m, const float* __restrict__ bn_r,
                        float* __restrict__ h) {
  const int lane = threadIdx.x;
  const int n0   = blockIdx.x * 16;   // 0..7 tiles
  const int m0   = blockIdx.y * 16;   // 0..48 tiles (784 rows exactly)
  const int hl   = lane >> 4;
  const int ll   = lane & 15;

  const int mrow = m0 + ll;           // global patch row 0..783
  const int b    = mrow / NPATCH;
  const int pi   = mrow % NPATCH;
  const int pr   = pi / 14, pc = pi % 14;
  const float* xbase = x + (size_t)b * 3 * 224 * 224;
  const float* Wrow  = pw + (size_t)(n0 + ll) * 768;
  __builtin_prefetch(Wrow, 0, 0);

  v8f acc = {};
  for (int kt = 0; kt < 24; ++kt) {
    v16h af, bf;
#pragma unroll
    for (int half = 0; half < 2; ++half) {
      const int kb  = kt * 32 + 8 * hl + 16 * half;
      const int c   = kb >> 8;
      const int rem = kb & 255;
      const int rr  = rem >> 4;
      const int cc  = rem & 15;     // 0 or 8: 8 contiguous floats in x
      const float* src = xbase + (size_t)c * 224 * 224 +
                         (size_t)(pr * 16 + rr) * 224 + pc * 16 + cc;
#pragma unroll
      for (int j = 0; j < 8; ++j) af[half * 8 + j] = (_Float16)src[j];
    }
    const int kB = kt * 32 + 8 * hl;
#pragma unroll
    for (int j = 0; j < 16; ++j) bf[j] = (_Float16)Wrow[kB + ((j < 8) ? j : 8 + j)];
    acc = wmma_f16(af, bf, acc);
  }

#pragma unroll
  for (int r = 0; r < 8; ++r) {
    const int m  = m0 + r + 8 * hl;       // output patch row
    const int b2 = m / NPATCH;
    const int p2 = m % NPATCH;
    const int n  = n0 + ll;
    float v = acc[r] + pb[n] + pos[(size_t)(1 + p2) * DIM + n];
    v = bn_g[n] * ((v - bn_m[n]) / (bn_r[n] + EPSV)) + bn_b[n];
    h[((size_t)b2 * LTOK + 1 + p2) * DIM + n] = v;
  }
}

__global__ void cls_init_kernel(const float* __restrict__ cls, const float* __restrict__ pos,
                                const float* __restrict__ bn_g, const float* __restrict__ bn_b,
                                const float* __restrict__ bn_m, const float* __restrict__ bn_r,
                                float* __restrict__ h) {
  const int idx = blockIdx.x * blockDim.x + threadIdx.x;
  if (idx >= BATCH * DIM) return;
  const int b = idx / DIM, d = idx % DIM;
  float v = cls[d] + pos[d];
  v = bn_g[d] * ((v - bn_m[d]) / (bn_r[d] + EPSV)) + bn_b[d];
  h[(size_t)b * LTOK * DIM + d] = v;
}

// ---------------------------------------------------------------------------
// LayerNorm over DIM=128 (one block per row, blockDim = 128).
// in row stride configurable (final LN reads strided cls rows).
// ---------------------------------------------------------------------------
__global__ __launch_bounds__(128)
void ln_kernel(const float* __restrict__ in, const float* __restrict__ w,
               const float* __restrict__ b, float* __restrict__ out, int in_stride) {
  __shared__ float red[128];
  const int row = blockIdx.x, t = threadIdx.x;
  const float v = in[(size_t)row * in_stride + t];
  red[t] = v;
  __syncthreads();
  for (int s = 64; s > 0; s >>= 1) { if (t < s) red[t] += red[t + s]; __syncthreads(); }
  const float mean = red[0] * (1.0f / 128.0f);
  __syncthreads();
  const float d = v - mean;
  red[t] = d * d;
  __syncthreads();
  for (int s = 64; s > 0; s >>= 1) { if (t < s) red[t] += red[t + s]; __syncthreads(); }
  const float var = red[0] * (1.0f / 128.0f);
  out[(size_t)row * DIM + t] = d * rsqrtf(var + EPSV) * w[t] + b[t];
}

// ---------------------------------------------------------------------------
// Tropical (max,+) linear: out[m,o] = max_k(x[m,k] + W[o,k]) + bias[o]. K=128.
// One block per row m; activation row staged in LDS; pure v_max_num_f32 loop.
// ---------------------------------------------------------------------------
__global__ __launch_bounds__(128)
void trop_lin_kernel(const float* __restrict__ X, const float* __restrict__ W,
                     const float* __restrict__ bias, float* __restrict__ out, int O) {
  __shared__ float xs[DIM];
  const int m = blockIdx.x, t = threadIdx.x;
  xs[t] = X[(size_t)m * DIM + t];
  __syncthreads();
  for (int o = t; o < O; o += 128) {
    const float* w = W + (size_t)o * DIM;
    float mv = -1e30f;
#pragma unroll 4
    for (int k = 0; k < DIM; ++k) mv = fmaxf(mv, xs[k] + w[k]);
    out[(size_t)m * O + o] = mv + bias[o];
  }
}

__global__ void gate_kernel(const float* __restrict__ hn, const float* __restrict__ gw,
                            const float* __restrict__ gb, float* __restrict__ g) {
  const int idx = blockIdx.x * blockDim.x + threadIdx.x;
  if (idx >= M_TOK * NH) return;
  const int m = idx / NH, hh = idx % NH;
  float s = gb[hh];
  for (int k = 0; k < DIM; ++k) s += hn[(size_t)m * DIM + k] * gw[(size_t)hh * DIM + k];
  g[idx] = 1.0f / (1.0f + __expf(-s));
}

// ---------------------------------------------------------------------------
// Fused attention per (m-tile, head, batch):
//   K staged via global_load_async_to_lds_b128 (ASYNCcnt), V converted via VGPRs,
//   cs via WMMA (Q.K^T, K-dim 16 zero-padded to 32), ts via VALU max-plus,
//   gated blend + temp, row softmax, attn @ V via WMMA split across 4 waves.
// blockDim = 128 (4 waves). ~46.5 KB LDS.
// ---------------------------------------------------------------------------
__global__ __launch_bounds__(128)
void attn_kernel(const float* __restrict__ Q, const float* __restrict__ Kc,
                 const float* __restrict__ V, const float* __restrict__ g,
                 const float* __restrict__ temp, float* __restrict__ out, int layer) {
  __shared__ float    Qf[16][DK];
  __shared__ float    Kf[LPAD][DK];
  __shared__ _Float16 Vh[VPAD][DK];
  __shared__ float    sc[16][LPAD];
  __shared__ _Float16 sch[16][VPAD];
  __shared__ float    red[16][8];
  __shared__ float    rowstat[16];
  __shared__ float    part[4][16][16];

  const int t = threadIdx.x, lane = t & 31, wave = t >> 5;
  const int hl = lane >> 4, ll = lane & 15;
  const int mtile = blockIdx.x, hh = blockIdx.y, b = blockIdx.z;
  const size_t base = (size_t)b * LTOK * DIM + (size_t)hh * DK;

  // K rows: async global->LDS DMA, 16 bytes per op (no VGPR round-trip; f32
  // kept for the tropical max-plus pass). Tracked by ASYNCcnt.
  for (int i = t; i < LTOK * 4; i += 128) {
    const int r = i >> 2, c4 = (i & 3) << 2;
    const unsigned lds = (unsigned)(size_t)&Kf[r][c4];
    const unsigned long long gp =
        (unsigned long long)(size_t)(Kc + base + (size_t)r * DIM + c4);
    asm volatile("global_load_async_to_lds_b128 %0, %1, off"
                 :: "v"(lds), "v"(gp) : "memory");
  }
  // Zero-fill K padding rows (197..207).
  for (int i = t; i < (LPAD - LTOK) * DK; i += 128)
    Kf[LTOK + i / DK][i % DK] = 0.0f;
  // V: f32 -> f16 conversion must pass through VGPRs; zero-pad to 224 rows.
  for (int i = t; i < VPAD * DK; i += 128) {
    const int r = i / DK, d = i % DK;
    Vh[r][d] = (_Float16)((r < LTOK) ? V[base + (size_t)r * DIM + d] : 0.0f);
  }
  for (int i = t; i < 16 * DK; i += 128) {
    const int r = i / DK, d = i % DK;
    const int row = mtile * 16 + r;
    Qf[r][d] = (row < LTOK) ? Q[base + (size_t)row * DIM + d] : 0.0f;
  }
  asm volatile("s_wait_asynccnt 0x0" ::: "memory");
  __syncthreads();

  // Classical scores cs: 13 key tiles split across the 4 waves.
  for (int nblk = wave; nblk < MT_L; nblk += 4) {
    v16h af = {}, bf = {};
#pragma unroll
    for (int j = 0; j < 16; ++j) {
      const int kd = ((j < 8) ? j : 16 + (j - 8)) + 8 * hl;
      if (kd < DK) af[j] = (_Float16)Qf[ll][kd];
    }
    const int ncol = nblk * 16 + ll;
#pragma unroll
    for (int j = 0; j < 16; ++j) {
      const int kd = ((j < 8) ? j : 16 + (j - 8)) + 8 * hl;
      if (kd < DK) bf[j] = (_Float16)Kf[ncol][kd];
    }
    v8f acc = {};
    acc = wmma_f16(af, bf, acc);
#pragma unroll
    for (int r = 0; r < 8; ++r) sc[r + 8 * hl][nblk * 16 + ll] = acc[r];
  }
  __syncthreads();

  // Tropical scores + gated blend + temperature.
  const float scale = 0.25f;                 // DK^-0.5
  const float tpv = temp[layer * NH + hh];
  for (int i = t; i < 16 * LTOK; i += 128) {
    const int q = i / LTOK, k = i % LTOK;
    float mv = -1e30f;
#pragma unroll
    for (int d = 0; d < DK; ++d) mv = fmaxf(mv, Qf[q][d] + Kf[k][d]);
    const int qrow = mtile * 16 + q;
    const float gg = (qrow < LTOK) ? g[((size_t)b * LTOK + qrow) * NH + hh] : 0.5f;
    sc[q][k] = (gg * (mv * scale) + (1.0f - gg) * (sc[q][k] * scale)) / tpv;
  }
  __syncthreads();

  // Row softmax over 197 keys: 8 threads per row.
  const int q = t >> 3, s = t & 7;
  float mval = -1e30f;
  for (int k = s; k < LTOK; k += 8) mval = fmaxf(mval, sc[q][k]);
  red[q][s] = mval;
  __syncthreads();
  if (s == 0) {
    float mm = red[q][0];
    for (int j = 1; j < 8; ++j) mm = fmaxf(mm, red[q][j]);
    rowstat[q] = mm;
  }
  __syncthreads();
  const float rmax = rowstat[q];
  float ssum = 0.0f;
  for (int k = s; k < LTOK; k += 8) {
    const float e = __expf(sc[q][k] - rmax);
    sc[q][k] = e;
    ssum += e;
  }
  red[q][s] = ssum;
  __syncthreads();
  if (s == 0) {
    float a = 0.0f;
    for (int j = 0; j < 8; ++j) a += red[q][j];
    rowstat[q] = a;
  }
  __syncthreads();
  const float inv = 1.0f / rowstat[q];
  for (int k = s; k < VPAD; k += 8)
    sch[q][k] = (_Float16)((k < LTOK) ? sc[q][k] * inv : 0.0f);
  __syncthreads();

  // attn @ V: K = 224 (7 steps of 32) split across the 4 waves; partial
  // 16x16 f32 tiles reduced through LDS.
  {
    v8f acc = {};
    for (int kt = wave; kt < 7; kt += 4) {
      v16h af, bf;
      const int ka = kt * 32 + 8 * hl;
#pragma unroll
      for (int j = 0; j < 8; ++j) af[j]     = sch[ll][ka + j];
#pragma unroll
      for (int j = 0; j < 8; ++j) af[8 + j] = sch[ll][ka + 16 + j];
      const int kB = kt * 32 + 8 * hl;
#pragma unroll
      for (int j = 0; j < 16; ++j) bf[j] = Vh[kB + ((j < 8) ? j : 8 + j)][ll];
      acc = wmma_f16(af, bf, acc);
    }
#pragma unroll
    for (int r = 0; r < 8; ++r) part[wave][r + 8 * hl][ll] = acc[r];
  }
  __syncthreads();
  for (int i = t; i < 256; i += 128) {
    const int qr = i >> 4, d = i & 15;
    const float v = part[0][qr][d] + part[1][qr][d] + part[2][qr][d] + part[3][qr][d];
    const int qrow = mtile * 16 + qr;
    if (qrow < LTOK)
      out[((size_t)b * LTOK + qrow) * DIM + (size_t)hh * DK + d] = v;
  }
}

// ---------------------------------------------------------------------------
// FFN fuse: P=8 max/min fold over affine images of z, blend with gelu branch.
// ---------------------------------------------------------------------------
__global__ void ffn_fuse_kernel(const float* __restrict__ z, const float* __restrict__ gelu_c,
                                const float* __restrict__ gf, const float* __restrict__ lf_a,
                                const float* __restrict__ lf_c, const float* __restrict__ lf_gate,
                                float* __restrict__ fused) {
  const int idx = blockIdx.x * blockDim.x + threadIdx.x;
  if (idx >= M_TOK * FF) return;
  const int f = idx % FF;
  const float zz = z[idx];
  float mx = -1e30f, mn = 1e30f;
#pragma unroll
  for (int p = 0; p < PP; ++p) {
    const float v = zz * lf_a[p * FF + f] + lf_c[p * FF + f];
    mx = fmaxf(mx, v);
    mn = fminf(mn, v);
  }
  const float gl = 1.0f / (1.0f + __expf(-lf_gate[f]));
  const float tr = gl * mx + (1.0f - gl) * mn;
  const float gg = gf[idx];
  fused[idx] = gg * tr + (1.0f - gg) * gelu_c[idx];
}

__global__ void head_kernel(const float* __restrict__ fn0, const float* __restrict__ hw,
                            const float* __restrict__ hb, float* __restrict__ out) {
  const int idx = blockIdx.x * blockDim.x + threadIdx.x;
  if (idx >= BATCH * NCLS) return;
  const int b = idx / NCLS, c = idx % NCLS;
  float s = hb[c];
  for (int k = 0; k < DIM; ++k) s += fn0[(size_t)b * DIM + k] * hw[(size_t)c * DIM + k];
  out[idx] = s;
}

// ---------------------------------------------------------------------------
extern "C" void kernel_launch(void* const* d_in, const int* in_sizes, int n_in,
                              void* d_out, int out_size, void* d_ws, size_t ws_size,
                              hipStream_t stream) {
  (void)in_sizes; (void)n_in; (void)out_size; (void)ws_size;
  const float* x       = (const float*)d_in[0];
  const float* patch_w = (const float*)d_in[1];
  const float* patch_b = (const float*)d_in[2];
  const float* cls_tok = (const float*)d_in[3];
  const float* pos     = (const float*)d_in[4];
  const float* bn_g    = (const float*)d_in[5];
  const float* bn_b    = (const float*)d_in[6];
  const float* bn_m    = (const float*)d_in[7];
  const float* bn_r    = (const float*)d_in[8];
  const float* n1w     = (const float*)d_in[9];
  const float* n1b     = (const float*)d_in[10];
  const float* n2w     = (const float*)d_in[11];
  const float* n2b     = (const float*)d_in[12];
  const float* wq      = (const float*)d_in[13];
  const float* bq      = (const float*)d_in[14];
  const float* wk      = (const float*)d_in[15];
  const float* bk      = (const float*)d_in[16];
  const float* wv      = (const float*)d_in[17];
  const float* bv      = (const float*)d_in[18];
  const float* wo      = (const float*)d_in[19];
  const float* bo      = (const float*)d_in[20];
  const float* gate_w  = (const float*)d_in[21];
  const float* gate_b  = (const float*)d_in[22];
  const float* temp    = (const float*)d_in[23];
  const float* tu_w    = (const float*)d_in[24];
  const float* tu_b    = (const float*)d_in[25];
  const float* lf_a    = (const float*)d_in[26];
  const float* lf_c    = (const float*)d_in[27];
  const float* lf_gate = (const float*)d_in[28];
  const float* cu_w    = (const float*)d_in[29];
  const float* cu_b    = (const float*)d_in[30];
  const float* fg_w    = (const float*)d_in[31];
  const float* fg_b    = (const float*)d_in[32];
  const float* dn_w    = (const float*)d_in[33];
  const float* dn_b    = (const float*)d_in[34];
  const float* fn_w    = (const float*)d_in[35];
  const float* fn_b    = (const float*)d_in[36];
  const float* head_w  = (const float*)d_in[37];
  const float* head_b  = (const float*)d_in[38];
  float* out = (float*)d_out;

  float* ws = (float*)d_ws;
  size_t off = 0;
  auto alloc = [&](size_t n) { float* p = ws + off; off += n; return p; };
  float* h    = alloc((size_t)M_TOK * DIM);
  float* hn   = alloc((size_t)M_TOK * DIM);
  float* Qb   = alloc((size_t)M_TOK * DIM);
  float* Kb   = alloc((size_t)M_TOK * DIM);
  float* Vb   = alloc((size_t)M_TOK * DIM);
  float* att  = alloc((size_t)M_TOK * DIM);
  float* gbuf = alloc((size_t)M_TOK * NH);
  float* zbuf = alloc((size_t)M_TOK * FF);
  float* cbuf = alloc((size_t)M_TOK * FF);
  float* gfb  = alloc((size_t)M_TOK * FF);
  float* fusd = alloc((size_t)M_TOK * FF);
  float* fn0  = alloc((size_t)BATCH * DIM);

  const int MTILES = (M_TOK + 15) / 16;   // 50

  // Embedding
  patch_embed_kernel<<<dim3(DIM / 16, 784 / 16), 32, 0, stream>>>(
      x, patch_w, patch_b, pos, bn_g, bn_b, bn_m, bn_r, h);
  cls_init_kernel<<<(BATCH * DIM + 127) / 128, 128, 0, stream>>>(
      cls_tok, pos, bn_g, bn_b, bn_m, bn_r, h);

  for (int l = 0; l < NLAYER; ++l) {
    // Attention block
    ln_kernel<<<M_TOK, 128, 0, stream>>>(h, n1w + l * DIM, n1b + l * DIM, hn, DIM);
    trop_lin_kernel<<<M_TOK, 128, 0, stream>>>(hn, wq + (size_t)l * DIM * DIM, bq + l * DIM, Qb, DIM);
    trop_lin_kernel<<<M_TOK, 128, 0, stream>>>(hn, wk + (size_t)l * DIM * DIM, bk + l * DIM, Kb, DIM);
    gemm_wmma_kernel<<<dim3(DIM / 16, MTILES), 32, 0, stream>>>(
        hn, wv + (size_t)l * DIM * DIM, bv + l * DIM, nullptr, Vb, M_TOK, DIM, DIM, 0);
    gate_kernel<<<(M_TOK * NH + 127) / 128, 128, 0, stream>>>(
        hn, gate_w + (size_t)l * NH * DIM, gate_b + l * NH, gbuf);
    attn_kernel<<<dim3(MT_L, NH, BATCH), 128, 0, stream>>>(Qb, Kb, Vb, gbuf, temp, att, l);
    gemm_wmma_kernel<<<dim3(DIM / 16, MTILES), 32, 0, stream>>>(
        att, wo + (size_t)l * DIM * DIM, bo + l * DIM, h, h, M_TOK, DIM, DIM, 0);

    // FFN block
    ln_kernel<<<M_TOK, 128, 0, stream>>>(h, n2w + l * DIM, n2b + l * DIM, hn, DIM);
    trop_lin_kernel<<<M_TOK, 128, 0, stream>>>(hn, tu_w + (size_t)l * FF * DIM, tu_b + l * FF, zbuf, FF);
    gemm_wmma_kernel<<<dim3(FF / 16, MTILES), 32, 0, stream>>>(
        hn, cu_w + (size_t)l * FF * DIM, cu_b + l * FF, nullptr, cbuf, M_TOK, FF, DIM, 1);
    gemm_wmma_kernel<<<dim3(FF / 16, MTILES), 32, 0, stream>>>(
        hn, fg_w + (size_t)l * FF * DIM, fg_b + l * FF, nullptr, gfb, M_TOK, FF, DIM, 2);
    ffn_fuse_kernel<<<(M_TOK * FF + 255) / 256, 256, 0, stream>>>(
        zbuf, cbuf, gfb, lf_a + (size_t)l * PP * FF, lf_c + (size_t)l * PP * FF,
        lf_gate + (size_t)l * FF, fusd);
    gemm_wmma_kernel<<<dim3(DIM / 16, MTILES), 32, 0, stream>>>(
        fusd, dn_w + (size_t)l * DIM * FF, dn_b + l * DIM, h, h, M_TOK, DIM, FF, 0);
  }

  // Final LN on cls rows (stride L*D) + classifier head.
  ln_kernel<<<BATCH, 128, 0, stream>>>(h, fn_w, fn_b, fn0, LTOK * DIM);
  head_kernel<<<(BATCH * NCLS + 255) / 256, 256, 0, stream>>>(fn0, head_w, head_b, out);
}
